// RNN_9775345566020
// MI455X (gfx1250) — compile-verified, matrix operand order checked by reading, not run on hardware
//
#include <hip/hip_runtime.h>

// ---------------- problem constants (from reference) ----------------
#define T_STEPS 500
#define BATCH   128
#define IN_DIM  128
#define N_HID   1024
#define O_DIM   2
#define DELAY   5
#define ALPHA   0.1f
#define ONE_MINUS_ALPHA 0.9f
#define FB_DENSITY 0.1f

#define NWG        64      // persistent grid: 64 WGs x 256 threads = 512 waves
#define THREADS    256
#define WAVES_PER_WG 8
#define GEMM_WGS   32      // WGs 0..31: hidden-state GEMM (256 waves, 16x32 tiles)
                           // WGs 32..63: readout dot products (256 waves, 1 per (b,o))

// WMMA vector types (probe-verified signatures)
typedef __bf16 v16bf __attribute__((ext_vector_type(16)));
typedef __bf16 v8bf  __attribute__((ext_vector_type(8)));
typedef float  v8f   __attribute__((ext_vector_type(8)));

// ---------------- workspace layout (bytes) ----------------
// 0      : barrier {count, gen} (2 x u32), padded to 256
// 256    : Whh  bf16  [1024 x 1024]       = 2 MB
// +      : Wih  bf16  [1024 x 128]        = 256 KB
// +      : r1   bf16  [2][128 x 1024]     = 512 KB  (ping-pong A-operand buffers)
// +      : x    f32   [128 x 1024]        = 512 KB  (leaky-integrator state)
static constexpr size_t WS_BAR  = 0;
static constexpr size_t WS_WHH  = 256;
static constexpr size_t WS_WIH  = WS_WHH + (size_t)N_HID * N_HID * 2;
static constexpr size_t WS_R1   = WS_WIH + (size_t)N_HID * IN_DIM * 2;
static constexpr size_t WS_X    = WS_R1  + (size_t)2 * BATCH * N_HID * 2;

// ---------------- grid-wide sense-reversing barrier ----------------
__device__ __forceinline__ void grid_barrier(unsigned* cnt, unsigned* gen) {
    __threadfence();          // publish this WG's global writes
    __syncthreads();
    if (threadIdx.x == 0) {
        volatile unsigned* vgen = gen;
        unsigned g = *vgen;                       // read generation BEFORE arriving
        unsigned arrived = atomicAdd(cnt, 1u) + 1u;
        if (arrived == NWG) {
            atomicExch(cnt, 0u);                  // reset for next barrier
            __threadfence();
            atomicAdd(gen, 1u);                   // release everyone
        } else {
            while (*vgen == g) { __builtin_amdgcn_s_sleep(1); }
        }
    }
    __syncthreads();
    __threadfence();
}

// ---------------- init kernel (stream-ordered before persistent kernel) ----------
__global__ void rnn_init_kernel(const float* __restrict__ x0_rand,
                                const float* __restrict__ Whh,
                                const float* __restrict__ Wih,
                                float*  __restrict__ poserr,      // [T+1, B, O]
                                float*  __restrict__ hidden,      // [T+1, B, N]
                                __bf16* __restrict__ whh_bf,
                                __bf16* __restrict__ wih_bf,
                                __bf16* __restrict__ r1_bf,       // parity-0 buffer
                                float*  __restrict__ x_ws,
                                unsigned* __restrict__ bar) {
    const int tid    = blockIdx.x * blockDim.x + threadIdx.x;
    const int stride = gridDim.x * blockDim.x;

    if (tid == 0) { bar[0] = 0u; bar[1] = 0u; }   // d_ws is poisoned 0xAA -> must zero

    // poserr row 0 = zeros (also serves as the pre-t=0 delayed error)
    for (int i = tid; i < BATCH * O_DIM; i += stride) poserr[i] = 0.0f;

    // x0 = (rand-0.5)*0.2 ; r0 = relu(x0) ; seed state + hidden[0]
    for (int i = tid; i < BATCH * N_HID; i += stride) {
        float x0 = (x0_rand[i] - 0.5f) * 0.2f;
        float r0 = fmaxf(x0, 0.0f);
        x_ws[i]   = x0;
        r1_bf[i]  = (__bf16)r0;                   // step-0 A operand (parity 0)
        hidden[i] = r0;
    }
    // bf16 weight copies (stay resident in the 192 MB L2 for the whole run)
    for (int i = tid; i < N_HID * N_HID; i += stride) whh_bf[i] = (__bf16)Whh[i];
    for (int i = tid; i < N_HID * IN_DIM; i += stride) wih_bf[i] = (__bf16)Wih[i];
}

// ---------------- persistent RNN kernel ----------------
__global__ __launch_bounds__(THREADS, 1)
void rnn_persistent_kernel(const float* __restrict__ X,        // [T,B,I]
                           const float* __restrict__ Xpert,    // [T,B,O]
                           const float* __restrict__ Xref,     // [T,B,O]
                           const float* __restrict__ Wout,     // [O,N]
                           const float* __restrict__ b_out,    // [O]
                           const float* __restrict__ Wfb,      // [N,O]
                           const float* __restrict__ b_fb,     // [N]
                           const float* __restrict__ mask_raw, // [N,O]
                           float*  __restrict__ poserr,        // [T+1,B,O]
                           float*  __restrict__ hidden,        // [T+1,B,N]
                           const __bf16* __restrict__ whh_bf,  // [N,N] (n-major, k contiguous)
                           const __bf16* __restrict__ wih_bf,  // [N,I]
                           __bf16* __restrict__ r1_bf,         // [2][B,N] ping-pong
                           float*  __restrict__ x_ws,          // [B,N]
                           unsigned* __restrict__ bar) {
    const int lane  = threadIdx.x & 31;
    const int wave  = threadIdx.x >> 5;
    const int gwave = blockIdx.x * WAVES_PER_WG + wave;   // 0..511
    const bool is_gemm = (blockIdx.x < GEMM_WGS);

    const int col  = lane & 15;   // A: tile row; B/C/D: tile column
    const int half = lane >> 4;   // K sub-block / M sub-block selector

    // ---- GEMM-wave invariants: 16(M) x 32(N) tile per wave, gwave 0..255 ----
    const int m0 = (gwave >> 5) * 16;        // 8 M-tiles
    const int n0 = (gwave & 31) * 32;        // 32 N-pairs
    const int nA = n0 + col;                 // first  output column for this lane
    const int nB = n0 + 16 + col;            // second output column for this lane

    float weff0A = 0.f, weff1A = 0.f, bfbA = 0.f;
    float weff0B = 0.f, weff1B = 0.f, bfbB = 0.f;
    const __bf16 *BrowA = nullptr, *BrowB = nullptr, *BIrowA = nullptr, *BIrowB = nullptr;
    if (is_gemm) {
        weff0A = (mask_raw[nA * O_DIM + 0] < FB_DENSITY) ? Wfb[nA * O_DIM + 0] : 0.0f;
        weff1A = (mask_raw[nA * O_DIM + 1] < FB_DENSITY) ? Wfb[nA * O_DIM + 1] : 0.0f;
        weff0B = (mask_raw[nB * O_DIM + 0] < FB_DENSITY) ? Wfb[nB * O_DIM + 0] : 0.0f;
        weff1B = (mask_raw[nB * O_DIM + 1] < FB_DENSITY) ? Wfb[nB * O_DIM + 1] : 0.0f;
        bfbA = b_fb[nA];
        bfbB = b_fb[nB];
        BrowA  = whh_bf + (size_t)nA * N_HID;    // Whh[n,:] -> B[k,col], k contiguous
        BrowB  = whh_bf + (size_t)nB * N_HID;
        BIrowA = wih_bf + (size_t)nA * IN_DIM;
        BIrowB = wih_bf + (size_t)nB * IN_DIM;
    }

    // ---- readout-wave invariants: one (b,o) dot product per wave ----
    const int w2 = gwave - 256;                       // 0..255 for readout WGs
    const int rb = (w2 >> 1) & (BATCH - 1);
    const int ro = w2 & 1;
    const float* wout_row = Wout + (size_t)ro * N_HID;

    // One barrier per epoch. Epoch j: GEMM waves run step j (j<T) while
    // readout waves run step j-1 (j>0) -- only coupling is the 5-step delay.
    for (int j = 0; j <= T_STEPS; ++j) {
        if (is_gemm && j < T_STEPS) {
            const __bf16* r_in  = r1_bf + (size_t)(j & 1)       * BATCH * N_HID;
            __bf16*       r_out = r1_bf + (size_t)((j + 1) & 1) * BATCH * N_HID;
            const __bf16* Arow  = r_in + (size_t)(m0 + col) * N_HID;

            v8f acc0 = {}, acc1 = {};

            // in2 = r1 @ Whh.T  (K=1024, 64 WMMAs; A reused across both N-halves)
            #pragma unroll 2
            for (int kt = 0; kt < N_HID; kt += 32) {
                const int kb = kt + half * 8;
                v8bf a_lo = *(const v8bf*)(Arow + kb);
                v8bf a_hi = *(const v8bf*)(Arow + kb + 16);
                v8bf b0lo = *(const v8bf*)(BrowA + kb);
                v8bf b0hi = *(const v8bf*)(BrowA + kb + 16);
                v8bf b1lo = *(const v8bf*)(BrowB + kb);
                v8bf b1hi = *(const v8bf*)(BrowB + kb + 16);
                v16bf a, b0, b1;
                #pragma unroll
                for (int i = 0; i < 8; ++i) {
                    a[i]  = a_lo[i]; a[i + 8]  = a_hi[i];
                    b0[i] = b0lo[i]; b0[i + 8] = b0hi[i];
                    b1[i] = b1lo[i]; b1[i + 8] = b1hi[i];
                }
                acc0 = __builtin_amdgcn_wmma_f32_16x16x32_bf16(
                    false, a, false, b0, (short)0, acc0, false, false);
                acc1 = __builtin_amdgcn_wmma_f32_16x16x32_bf16(
                    false, a, false, b1, (short)0, acc1, false, false);
            }

            // in1 = X[t] @ Wih.T (K=128, 8 WMMAs; A converted f32->bf16 in regs)
            const float* Xrow = X + (size_t)j * BATCH * IN_DIM
                                  + (size_t)(m0 + col) * IN_DIM;
            #pragma unroll
            for (int kt = 0; kt < IN_DIM; kt += 32) {
                const int kb = kt + half * 8;
                v16bf a, b0, b1;
                #pragma unroll
                for (int i = 0; i < 8; ++i) {
                    a[i]     = (__bf16)Xrow[kb + i];
                    a[i + 8] = (__bf16)Xrow[kb + 16 + i];
                }
                v8bf b0lo = *(const v8bf*)(BIrowA + kb);
                v8bf b0hi = *(const v8bf*)(BIrowA + kb + 16);
                v8bf b1lo = *(const v8bf*)(BIrowB + kb);
                v8bf b1hi = *(const v8bf*)(BIrowB + kb + 16);
                #pragma unroll
                for (int i = 0; i < 8; ++i) {
                    b0[i] = b0lo[i]; b0[i + 8] = b0hi[i];
                    b1[i] = b1lo[i]; b1[i + 8] = b1hi[i];
                }
                acc0 = __builtin_amdgcn_wmma_f32_16x16x32_bf16(
                    false, a, false, b0, (short)0, acc0, false, false);
                acc1 = __builtin_amdgcn_wmma_f32_16x16x32_bf16(
                    false, a, false, b1, (short)0, acc1, false, false);
            }

            // fused epilogue: + err@Weff + b_fb, leaky integrate, ReLU, store
            int erow = j - DELAY; if (erow < 0) erow = 0;    // poserr[0] == 0
            const float* errb = poserr + (size_t)erow * BATCH * O_DIM;
            float* hidrow = hidden + (size_t)(j + 1) * BATCH * N_HID;
            #pragma unroll
            for (int g = 0; g < 8; ++g) {
                const int b = m0 + g + 8 * half;             // lanes 16-31 -> M+8
                const float e0 = errb[b * O_DIM + 0];
                const float e1 = errb[b * O_DIM + 1];
                const size_t rowb = (size_t)b * N_HID;

                const float totA = acc0[g] + e0 * weff0A + e1 * weff1A + bfbA;
                const float xnA  = ONE_MINUS_ALPHA * x_ws[rowb + nA] + ALPHA * totA;
                const float rrA  = fmaxf(xnA, 0.0f);
                x_ws[rowb + nA]   = xnA;
                r_out[rowb + nA]  = (__bf16)rrA;
                hidrow[rowb + nA] = rrA;

                const float totB = acc1[g] + e0 * weff0B + e1 * weff1B + bfbB;
                const float xnB  = ONE_MINUS_ALPHA * x_ws[rowb + nB] + ALPHA * totB;
                const float rrB  = fmaxf(xnB, 0.0f);
                x_ws[rowb + nB]   = xnB;
                r_out[rowb + nB]  = (__bf16)rrB;
                hidrow[rowb + nB] = rrB;
            }
        } else if (!is_gemm && j > 0) {
            // readout for step jj = j-1: vt = r1 @ Wout.T + b_out ; v1 = ref-(vt+pert)
            const int jj = j - 1;
            const float* hrow = hidden + (size_t)(jj + 1) * BATCH * N_HID
                                       + (size_t)rb * N_HID;
            float s = 0.0f;
            #pragma unroll 8
            for (int i = lane; i < N_HID; i += 32) s += hrow[i] * wout_row[i];
            #pragma unroll
            for (int off = 16; off > 0; off >>= 1) s += __shfl_xor(s, off, 32);
            if (lane == 0) {
                const float vt = s + b_out[ro];
                const size_t io = (size_t)jj * BATCH * O_DIM + (size_t)rb * O_DIM + ro;
                poserr[(size_t)(jj + 1) * BATCH * O_DIM + (size_t)rb * O_DIM + ro] =
                    Xref[io] - (vt + Xpert[io]);
            }
        }

        grid_barrier(bar, bar + 1);     // single barrier per epoch
    }
}

// ---------------- launch ----------------
extern "C" void kernel_launch(void* const* d_in, const int* in_sizes, int n_in,
                              void* d_out, int out_size, void* d_ws, size_t ws_size,
                              hipStream_t stream) {
    (void)in_sizes; (void)n_in; (void)out_size; (void)ws_size;

    const float* X        = (const float*)d_in[0];
    const float* Xpert    = (const float*)d_in[1];
    const float* Xref     = (const float*)d_in[2];
    const float* x0_rand  = (const float*)d_in[3];
    const float* Wih      = (const float*)d_in[4];
    const float* Whh      = (const float*)d_in[5];
    const float* Wout     = (const float*)d_in[6];
    const float* b_out    = (const float*)d_in[7];
    const float* Wfb      = (const float*)d_in[8];
    const float* b_fb     = (const float*)d_in[9];
    const float* mask_raw = (const float*)d_in[10];

    float* poserr = (float*)d_out;                                        // [T+1,B,O]
    float* hidden = poserr + (size_t)(T_STEPS + 1) * BATCH * O_DIM;       // [T+1,B,N]

    char* ws = (char*)d_ws;
    unsigned* bar    = (unsigned*)(ws + WS_BAR);
    __bf16*   whh_bf = (__bf16*)(ws + WS_WHH);
    __bf16*   wih_bf = (__bf16*)(ws + WS_WIH);
    __bf16*   r1_bf  = (__bf16*)(ws + WS_R1);
    float*    x_ws   = (float*)(ws + WS_X);

    rnn_init_kernel<<<512, THREADS, 0, stream>>>(
        x0_rand, Whh, Wih, poserr, hidden, whh_bf, wih_bf, r1_bf, x_ws, bar);

    rnn_persistent_kernel<<<NWG, THREADS, 0, stream>>>(
        X, Xpert, Xref, Wout, b_out, Wfb, b_fb, mask_raw,
        poserr, hidden, whh_bf, wih_bf, r1_bf, x_ws, bar);
}
